// TorchCausalGroupedQueryPageAttention_55482387530054
// MI455X (gfx1250) — matrix operand order repaired and checked by reading, not compiled, not run on hardware
//
#include <hip/hip_runtime.h>

typedef __attribute__((ext_vector_type(16))) __bf16 v16bf;
typedef __attribute__((ext_vector_type(8)))  float  v8f;
typedef __attribute__((ext_vector_type(4)))  unsigned int u32x4;
typedef __attribute__((ext_vector_type(8)))  int    i32x8;
typedef __attribute__((ext_vector_type(4)))  int    i32x4;

#define NB    4                    // sequences
#define QL    512                  // new query tokens per seq
#define KVL   2048                 // kv context per seq
#define PASTL (KVL - QL)           // 1536
#define NQO   32
#define NKV   8
#define GQA   (NQO / NKV)          // 4
#define HD    128
#define PBS   16                   // page size
#define BPS   (KVL / PBS)          // 128 blocks per seq
#define QT    64                   // q rows per workgroup (4 waves x 16)
#define KVROW (NKV * HD)           // 1024-element row stride, same for cache & new
#define SM_SCALE 0.08838834764831845f   // 1/sqrt(128)

static __device__ __forceinline__ v16bf ld16pair(const __bf16* p0, const __bf16* p1) {
  v16bf r;
#pragma unroll
  for (int i = 0; i < 8; ++i) { r[i] = p0[i]; r[8 + i] = p1[i]; }
  return r;
}

// ---- TDM gather: 16 rows x 128 f32 from a [nrows x 1024]-strided tensor ----
// Rows are base..base+15 (paged-cache pages and new-token runs are both
// 16 consecutive rows of their source tensor). Descriptor per ISA 08 §8.3-8.7.
static __device__ __forceinline__ void tdm_gather16(uint32_t lds_off,
                                                    const float* gbase,
                                                    int rowbase, int nrows) {
  const uint64_t ga = (uint64_t)(uintptr_t)gbase;
  u32x4 g0;
  g0[0] = 1u | (1u << 31);                 // count=1, 16-bit idx, gather_mode=1
  g0[1] = lds_off;                         // lds_addr (bytes)
  g0[2] = (unsigned)(ga & 0xffffffffu);    // global_addr[31:0]
  g0[3] = (unsigned)((ga >> 32) & 0x1ffffffu) | (2u << 30);  // addr[56:32],type=2

  i32x8 g1;
  g1[0] = 2 << 16;                         // workgroup_mask=0, data_size=4B
  g1[1] = HD << 16;                        // tensor_dim0 = 128 (bits 79:48)
  g1[2] = (int)(((unsigned)nrows & 0xffffu) << 16);   // tensor_dim1[15:0]
  g1[3] = (int)((((unsigned)nrows >> 16) & 0xffffu) | ((unsigned)HD << 16)); // dim1 hi | tile_dim0=128
  g1[4] = 16;                              // tile_dim1 = 16 gather indices
  g1[5] = KVROW;                           // tensor_dim0_stride = 1024 elements
  g1[6] = 0;
  g1[7] = 0;

  i32x4 g2, g3;                            // packed 16-bit row indices
#pragma unroll
  for (int j = 0; j < 4; ++j) {
    const int r0 = rowbase + 2 * j;
    const int r8 = rowbase + 8 + 2 * j;
    g2[j] = (r0 & 0xffff) | ((r0 + 1) << 16);
    g3[j] = (r8 & 0xffff) | ((r8 + 1) << 16);
  }
  i32x8 gx;                                // 6-arg toolchain form: zero-fill
#pragma unroll
  for (int j = 0; j < 8; ++j) gx[j] = 0;
  __builtin_amdgcn_tensor_load_to_lds(g0, g1, g2, g3, gx, 0);
}

__global__ __launch_bounds__(128)
void paged_gqa_flash_kernel(const float* __restrict__ q,
                            const float* __restrict__ knew,
                            const float* __restrict__ vnew,
                            const float* __restrict__ kcache,
                            const float* __restrict__ vcache,
                            const int*   __restrict__ btab,
                            float* __restrict__ out)
{
  // Double-buffered raw f32 tiles filled by the Tensor Data Mover
  __shared__ __align__(32) float  Kf32[2][32][HD];   // 32 KB
  __shared__ __align__(32) float  Vf32[2][32][HD];   // 32 KB
  // bf16 working tiles
  __shared__ __align__(32) __bf16 Klds[32][HD];      // 8 KB  (token x d)
  __shared__ __align__(32) __bf16 Vt[HD][40];        // 10 KB (d x token, padded)
  __shared__ __align__(32) float  Sbuf[4][16][32];   // 8 KB, per-wave S tile
  __shared__ __align__(32) __bf16 Pbuf[4][16][32];   // 4 KB, per-wave P tile

  const int qt  = blockIdx.x;          // q tile
  const int qh  = blockIdx.y;          // query head
  const int b   = blockIdx.z;          // sequence
  const int hkv = qh / GQA;            // kv head
  const int q0  = qt * QT;

  const int tid  = threadIdx.x;
  const int w_s  = __builtin_amdgcn_readfirstlane(tid) >> 5;  // scalar wave id
  const int w    = tid >> 5;
  const int lane = tid & 31;
  const int mrw  = lane & 15;
  const int hi   = lane >> 4;
  const int qrow = q0 + 16 * w + mrw;

  const int kv_limit = min(KVL, q0 + QT + PASTL);
  const int ntiles   = kv_limit >> 5;   // kv_limit is a multiple of 32

  // ---- issue TDM gathers for one 32-token tile into buffer `buf` ----
  auto prefetch = [&](int n, int buf) {
#pragma unroll
    for (int gci = 0; gci < 2; ++gci) {
      const int t0 = 32 * n + 16 * gci;
      int rowbase, nrows;
      const float* ks;
      const float* vs;
      if (t0 >= PASTL) {                 // new tokens: contiguous rows of knew
        rowbase = b * QL + (t0 - PASTL);
        nrows   = NB * QL;
        ks = knew; vs = vnew;
      } else {                           // old tokens: one cache page
        const int blk = __builtin_amdgcn_readfirstlane(btab[b * BPS + (t0 >> 4)]);
        rowbase = blk * PBS;
        nrows   = NB * BPS * PBS;        // 8192 cache rows
        ks = kcache; vs = vcache;
      }
      tdm_gather16((uint32_t)(uintptr_t)&Kf32[buf][16 * gci][0],
                   ks + hkv * HD, rowbase, nrows);
      tdm_gather16((uint32_t)(uintptr_t)&Vf32[buf][16 * gci][0],
                   vs + hkv * HD, rowbase, nrows);
    }
  };

  // ---- Load Q (A-matrix layout, pre-scaled, bf16), 4 chunks of K-dim 32 ----
  const float* qp = q + ((size_t)(b * QL + qrow)) * (NQO * HD) + (size_t)qh * HD;
  const int abase = 8 * hi;
  v16bf qa[4];
#pragma unroll
  for (int c = 0; c < 4; ++c) {
    const int d0 = 32 * c;
#pragma unroll
    for (int j = 0; j < 8; ++j) {
      qa[c][j]     = (__bf16)(qp[d0 + abase + j]      * SM_SCALE);
      qa[c][8 + j] = (__bf16)(qp[d0 + 16 + abase + j] * SM_SCALE);
    }
  }

  // ---- Flash state ----
  v8f acc[8];
#pragma unroll
  for (int c2 = 0; c2 < 8; ++c2)
#pragma unroll
    for (int r = 0; r < 8; ++r) acc[c2][r] = 0.0f;
  float mprev = -__builtin_inff();
  float lsum  = 0.0f;

  if (w_s == 0) prefetch(0, 0);        // uniform branch: wave 0 drives the TDM

  int buf = 0;
  for (int n = 0; n < ntiles; ++n) {
    const int kv0 = 32 * n;

    if (w_s == 0) __builtin_amdgcn_s_wait_tensorcnt((short)0);
    __syncthreads();                   // raw tile n visible; Klds/Vt reusable

    if (w_s == 0 && (n + 1) < ntiles) prefetch(n + 1, buf ^ 1);

    // ---- LDS->LDS convert: f32 tile -> bf16 K (row-major) + V (transposed) --
#pragma unroll 8
    for (int i = 0; i < 32; ++i) {
      Klds[i][tid] = (__bf16)Kf32[buf][i][tid];
      Vt[tid][i]   = (__bf16)Vf32[buf][i][tid];
    }
    __syncthreads();

    // ---- S = Q K^T for two 16-col halves (v_wmma_f32_16x16x32_bf16) ----
#pragma unroll
    for (int ch = 0; ch < 2; ++ch) {
      v8f sc;
#pragma unroll
      for (int r = 0; r < 8; ++r) sc[r] = 0.0f;
#pragma unroll
      for (int c = 0; c < 4; ++c) {
        const __bf16* kp = &Klds[16 * ch + mrw][32 * c + 8 * hi];
        v16bf kb = ld16pair(kp, kp + 16);
        sc = __builtin_amdgcn_wmma_f32_16x16x32_bf16(false, qa[c], false, kb,
                                                     (short)0, sc, false, false);
      }
#pragma unroll
      for (int r = 0; r < 8; ++r)
        Sbuf[w][r + 8 * hi][16 * ch + mrw] = sc[r];
    }
    asm volatile("s_wait_dscnt 0" ::: "memory");

    // ---- Online softmax; mask only on boundary tiles (wave-uniform test) ----
    const bool need_mask =
        (kv0 + 31) - (q0 + 16 * w_s) > PASTL;   // scalar: any lane/col masked?
    float sv[16];
    float tmax = -__builtin_inff();
    if (need_mask) {
#pragma unroll
      for (int j = 0; j < 16; ++j) {
        const int col = 16 * hi + j;
        float x = Sbuf[w][mrw][col];
        if ((kv0 + col) - qrow > PASTL) x = -__builtin_inff();
        sv[j] = x;
        tmax = fmaxf(tmax, x);
      }
    } else {
#pragma unroll
      for (int j = 0; j < 16; ++j) {
        sv[j] = Sbuf[w][mrw][16 * hi + j];
        tmax = fmaxf(tmax, sv[j]);
      }
    }
    tmax = fmaxf(tmax, __shfl_xor(tmax, 16));
    const float mnew  = fmaxf(mprev, tmax);
    const float alpha = __expf(mprev - mnew);
    float ps = 0.0f;
#pragma unroll
    for (int j = 0; j < 16; ++j) {
      const float p = __expf(sv[j] - mnew);
      ps += p;
      Pbuf[w][mrw][16 * hi + j] = (__bf16)p;
    }
    ps += __shfl_xor(ps, 16);
    lsum  = lsum * alpha + ps;
    mprev = mnew;

    // ---- Rescale O accumulators (alpha per row via lane broadcast) ----
#pragma unroll
    for (int r = 0; r < 8; ++r) {
      const float ar = __shfl(alpha, r + 8 * hi);
#pragma unroll
      for (int c2 = 0; c2 < 8; ++c2) acc[c2][r] *= ar;
    }

    asm volatile("s_wait_dscnt 0" ::: "memory");

    // ---- O += P V : A = P (16x32 bf16), B = V^T tile columns ----
    v16bf pa = ld16pair(&Pbuf[w][mrw][8 * hi], &Pbuf[w][mrw][16 + 8 * hi]);
#pragma unroll
    for (int c2 = 0; c2 < 8; ++c2) {
      const __bf16* vp = &Vt[16 * c2 + mrw][8 * hi];
      v16bf vb = ld16pair(vp, vp + 16);
      acc[c2] = __builtin_amdgcn_wmma_f32_16x16x32_bf16(false, pa, false, vb,
                                                        (short)0, acc[c2],
                                                        false, false);
    }
    buf ^= 1;
  }

  // ---- Epilogue: divide by l and store fp32 output ----
  const float lrec = 1.0f / lsum;
#pragma unroll
  for (int r = 0; r < 8; ++r) {
    const int   qr = q0 + 16 * w + r + 8 * hi;
    const float lr = __shfl(lrec, r + 8 * hi);
    float* op = out + ((size_t)(b * QL + qr)) * (NQO * HD)
                    + (size_t)qh * HD + mrw;
#pragma unroll
    for (int c2 = 0; c2 < 8; ++c2) op[16 * c2] = acc[c2][r] * lr;
  }
}

extern "C" void kernel_launch(void* const* d_in, const int* in_sizes, int n_in,
                              void* d_out, int out_size, void* d_ws, size_t ws_size,
                              hipStream_t stream) {
  const float* q      = (const float*)d_in[0];
  const float* knew   = (const float*)d_in[1];
  const float* vnew   = (const float*)d_in[2];
  const float* kcache = (const float*)d_in[3];
  const float* vcache = (const float*)d_in[4];
  const int*   btab   = (const int*)d_in[5];
  // d_in[6] (new_cache_slots) is implied by block_tables + position; unused.
  float* out = (float*)d_out;

  dim3 grid(QL / QT, NQO, NB);   // 8 x 32 x 4 = 1024 workgroups
  dim3 block(128);               // 4 waves (wave32)
  paged_gqa_flash_kernel<<<grid, block, 0, stream>>>(q, knew, vnew, kcache,
                                                     vcache, btab, out);
}
